// BILSTM_CRF_49486613185338
// MI455X (gfx1250) — matrix operand order repaired and checked by reading, not compiled, not run on hardware
//
#include <hip/hip_runtime.h>
#include <hip/hip_bf16.h>

// ---------------- constants (match reference) ----------------
#define LSEQ   4096
#define DEMB   256
#define HHID   256          // per-direction hidden
#define G4H    1024         // 4*H gate rows
#define DHID   512
#define NTAGS  24
#define TSTART 22
#define TSTOP  23
#define NEGINF -1.0e6f

// ---------------- WMMA vector types ----------------
typedef __attribute__((ext_vector_type(16))) _Float16 v16h;
typedef __attribute__((ext_vector_type(8)))  _Float16 v8h;
typedef __attribute__((ext_vector_type(8)))  float    v8f;

union AFrag { v16h v; v8h h[2]; };

// A fragment (16x32 f16, ISA 7.12.2): lane row m = lane&15,
//   lanes 0-15: K = kb+0..15 contiguous (this code's order, as B); lanes 16-31: K = kb+16..31.
// B fragment (32x16 f16): lane col n = lane&15,
//   lanes 0-15: K = kb+0..15 contiguous; lanes 16-31: K = kb+16..31.
// C/D (16x16 f32): vgpr r holds row m = r + 8*(lane>=16), col n = lane&15.

// ================= prep kernels =================
__global__ void k_gather_x(const int* __restrict__ sent,
                           const float* __restrict__ emb,
                           _Float16* __restrict__ X) {
  int i = blockIdx.x * blockDim.x + threadIdx.x;          // L*DEMB
  if (i >= LSEQ * DEMB) return;
  int t = i >> 8, d = i & 255;
  X[i] = (_Float16)emb[(size_t)sent[t] * DEMB + d];
}

__global__ void k_cvt_pair(const float* __restrict__ a,
                           const float* __restrict__ b,
                           _Float16* __restrict__ out, int n) {
  int i = blockIdx.x * blockDim.x + threadIdx.x;          // 2*n
  if (i >= 2 * n) return;
  out[i] = (_Float16)((i < n) ? a[i] : b[i - n]);
}

__global__ void k_cvt_wout(const float* __restrict__ w,
                           _Float16* __restrict__ out) {  // [32][512], zero pad
  int i = blockIdx.x * blockDim.x + threadIdx.x;
  if (i >= 32 * DHID) return;
  int n = i >> 9, k = i & 511;
  out[i] = (n < NTAGS) ? (_Float16)w[n * DHID + k] : (_Float16)0.f;
}

// ================= input projection GEMM (WMMA f16) =================
// XP[dir][t][4H] = x[(dir? L-1-t : t)] @ W_ih[dir].T + (b_ih + b_hh)
__global__ __launch_bounds__(256)
void k_gemm_inproj(const _Float16* __restrict__ X,     // [L][DEMB]
                   const _Float16* __restrict__ Wih,   // [2][4H][DEMB]
                   const float* __restrict__ bihf, const float* __restrict__ bhhf,
                   const float* __restrict__ bihb, const float* __restrict__ bhhb,
                   float* __restrict__ XP) {            // [2][L][4H]
  int wid  = (blockIdx.x * blockDim.x + threadIdx.x) >> 5;   // 32768 waves
  int lane = threadIdx.x & 31;
  int dir   = wid >> 14;
  int rem   = wid & 16383;
  int mtile = rem >> 6;              // 256 M-tiles
  int ntile = rem & 63;              // 64  N-tiles
  int rl = lane & 15, g = lane >> 4;
  int ncol = ntile * 16 + rl;
  int mrow = mtile * 16 + rl;
  int arow = dir ? (LSEQ - 1 - mrow) : mrow;
  const _Float16* xr = X + (size_t)arow * DEMB;
  const _Float16* w  = Wih + (size_t)dir * G4H * DEMB + (size_t)ncol * DEMB;
  v8f acc = {};
  for (int kb = 0; kb < DEMB; kb += 32) {
    AFrag a;
    a.h[0] = *(const v8h*)(xr + kb + g * 16);
    a.h[1] = *(const v8h*)(xr + kb + g * 16 + 8);
    v16h b = *(const v16h*)(w + kb + g * 16);
    acc = __builtin_amdgcn_wmma_f32_16x16x32_f16(false, a.v, false, b,
                                                 (short)0, acc, false, false);
  }
  float bias = dir ? (bihb[ncol] + bhhb[ncol]) : (bihf[ncol] + bhhf[ncol]);
  float* out = XP + (size_t)dir * LSEQ * G4H;
  int mbase = mtile * 16 + g * 8;
  for (int r = 0; r < 8; ++r)
    out[(size_t)(mbase + r) * G4H + ncol] = acc[r] + bias;
}

// ================= sequential LSTM scan (WMMA matvec, h broadcast) =========
// 2 blocks (dir 0/1) x 256 threads (8 waves). Each wave owns 8 row-tiles of
// W_hh (128 gate rows); h is replicated across all 16 B-columns from LDS.
__global__ __launch_bounds__(256)
void k_lstm_scan(const _Float16* __restrict__ Whh,  // [2][4H][H] f16
                 const float* __restrict__ XP,      // [2][L][4H]
                 const float* __restrict__ h0,      // [2][1][H]
                 const float* __restrict__ c0,      // [2][1][H]
                 _Float16* __restrict__ HS) {       // [L][DHID]
  __shared__ __align__(64) _Float16 hh[HHID];
  __shared__ float gates[G4H];
  int dir  = blockIdx.x;
  int tid  = threadIdx.x;
  int lane = tid & 31, wv = tid >> 5;
  int rl = lane & 15, g = lane >> 4;
  const _Float16* W  = Whh + (size_t)dir * G4H * HHID;
  const float*    xp = XP  + (size_t)dir * LSEQ * G4H;
  float c = c0[dir * HHID + tid];
  float h = h0[dir * HHID + tid];
  hh[tid] = (_Float16)h;
  __syncthreads();

  for (int t = 0; t < LSEQ; ++t) {
    // gates = W_hh @ h via 8 row-tiles/wave, K=256 in 8 WMMA steps
    for (int tile = 0; tile < 8; ++tile) {
      int row = (wv * 8 + tile) * 16 + rl;
      const _Float16* wr = W + (size_t)row * HHID;
      v8f acc = {};
      for (int kb = 0; kb < HHID; kb += 32) {
        AFrag a;
        a.h[0] = *(const v8h*)(wr + kb + g * 16);
        a.h[1] = *(const v8h*)(wr + kb + g * 16 + 8);
        v16h b = *(const v16h*)(&hh[kb + g * 16]);   // broadcast LDS read
        acc = __builtin_amdgcn_wmma_f32_16x16x32_f16(false, a.v, false, b,
                                                     (short)0, acc, false, false);
      }
      if (rl == 0) {                 // column 0 of D holds the matvec result
        int rb = (wv * 8 + tile) * 16 + g * 8;
        for (int r = 0; r < 8; ++r) gates[rb + r] = acc[r];
      }
    }
    __syncthreads();
    // cell update: thread tid owns hidden index tid
    {
      const float* xpt = xp + (size_t)t * G4H;
      float gi = gates[tid]           + xpt[tid];
      float gf = gates[HHID + tid]    + xpt[HHID + tid];
      float gg = gates[2*HHID + tid]  + xpt[2*HHID + tid];
      float go = gates[3*HHID + tid]  + xpt[3*HHID + tid];
      gi = 1.f / (1.f + __expf(-gi));
      gf = 1.f / (1.f + __expf(-gf));
      go = 1.f / (1.f + __expf(-go));
      gg = tanhf(gg);
      c = gf * c + gi * gg;
      h = go * tanhf(c);
      int trow = dir ? (LSEQ - 1 - t) : t;
      HS[(size_t)trow * DHID + dir * HHID + tid] = (_Float16)h;
      hh[tid] = (_Float16)h;
    }
    __syncthreads();
  }
}

// ================= output projection (WMMA f16) =================
__global__ __launch_bounds__(256)
void k_gemm_feats(const _Float16* __restrict__ HS,    // [L][DHID]
                  const _Float16* __restrict__ Wout,  // [32][DHID]
                  const float* __restrict__ bout,     // [NTAGS]
                  float* __restrict__ feats) {        // [L][NTAGS]
  int wid  = (blockIdx.x * blockDim.x + threadIdx.x) >> 5;   // 512 waves
  int lane = threadIdx.x & 31;
  int mtile = wid >> 1;
  int ntile = wid & 1;
  int rl = lane & 15, g = lane >> 4;
  int ncol = ntile * 16 + rl;
  const _Float16* xr = HS + (size_t)(mtile * 16 + rl) * DHID;
  const _Float16* w  = Wout + (size_t)ncol * DHID;
  v8f acc = {};
  for (int kb = 0; kb < DHID; kb += 32) {
    AFrag a;
    a.h[0] = *(const v8h*)(xr + kb + g * 16);
    a.h[1] = *(const v8h*)(xr + kb + g * 16 + 8);
    v16h b = *(const v16h*)(w + kb + g * 16);
    acc = __builtin_amdgcn_wmma_f32_16x16x32_f16(false, a.v, false, b,
                                                 (short)0, acc, false, false);
  }
  if (ncol < NTAGS) {
    float bias = bout[ncol];
    int mbase = mtile * 16 + g * 8;
    for (int r = 0; r < 8; ++r)
      feats[(size_t)(mbase + r) * NTAGS + ncol] = acc[r] + bias;
  }
}

// ================= Viterbi (single wave) =================
__global__ void k_viterbi(const float* __restrict__ feats,  // [L][NTAGS]
                          const float* __restrict__ trans,  // [NTAGS][NTAGS]
                          int* __restrict__ bptrs,          // [L][NTAGS]
                          float* __restrict__ out) {        // [1 + L]
  __shared__ float fv[NTAGS];
  __shared__ float tr[NTAGS * NTAGS];
  int lane = threadIdx.x;
  for (int i = lane; i < NTAGS * NTAGS; i += 32) tr[i] = trans[i];
  if (lane < NTAGS) fv[lane] = (lane == TSTART) ? 0.f : NEGINF;
  __syncthreads();
  for (int t = 0; t < LSEQ; ++t) {
    float best = -3.4e38f; int bp = 0;
    if (lane < NTAGS) {
      for (int p = 0; p < NTAGS; ++p) {
        float s = fv[p] + tr[lane * NTAGS + p];
        if (s > best) { best = s; bp = p; }
      }
      bptrs[t * NTAGS + lane] = bp;
    }
    __syncthreads();
    if (lane < NTAGS) fv[lane] = best + feats[(size_t)t * NTAGS + lane];
    __syncthreads();
  }
  if (lane == 0) {
    float bestS = -3.4e38f; int bt = 0;
    for (int p = 0; p < NTAGS; ++p) {
      float s = fv[p] + tr[TSTOP * NTAGS + p];
      if (s > bestS) { bestS = s; bt = p; }
    }
    out[0] = bestS;
    int tag = bt;
    out[1 + (LSEQ - 1)] = (float)tag;
    for (int t = LSEQ - 2; t >= 0; --t) {
      tag = bptrs[(t + 1) * NTAGS + tag];
      out[1 + t] = (float)tag;
    }
  }
}

// ================= launch =================
extern "C" void kernel_launch(void* const* d_in, const int* in_sizes, int n_in,
                              void* d_out, int out_size, void* d_ws, size_t ws_size,
                              hipStream_t stream) {
  (void)in_sizes; (void)n_in; (void)out_size; (void)ws_size;
  const int*   sent  = (const int*)  d_in[0];
  const float* h0    = (const float*)d_in[1];
  const float* c0    = (const float*)d_in[2];
  const float* emb   = (const float*)d_in[3];
  const float* Wihf  = (const float*)d_in[4];
  const float* Whhf  = (const float*)d_in[5];
  const float* bihf  = (const float*)d_in[6];
  const float* bhhf  = (const float*)d_in[7];
  const float* Wihb  = (const float*)d_in[8];
  const float* Whhb  = (const float*)d_in[9];
  const float* bihb  = (const float*)d_in[10];
  const float* bhhb  = (const float*)d_in[11];
  const float* Wout  = (const float*)d_in[12];
  const float* bout  = (const float*)d_in[13];
  const float* trans = (const float*)d_in[14];

  char* ws = (char*)d_ws;
  float*     XP     = (float*)    (ws + 0);                       // 32 MB
  _Float16*  X16    = (_Float16*) (ws + 33554432);                // 2 MB
  _Float16*  Wih16  = (_Float16*) (ws + 35651584);                // 1 MB
  _Float16*  Whh16  = (_Float16*) (ws + 36700160);                // 1 MB
  _Float16*  Wout16 = (_Float16*) (ws + 37748736);                // 32 KB
  _Float16*  HS     = (_Float16*) (ws + 37781504);                // 4 MB
  float*     feats  = (float*)    (ws + 41975808);                // 384 KB
  int*       bptrs  = (int*)      (ws + 42369024);                // 384 KB

  // prep
  k_gather_x<<<(LSEQ * DEMB + 255) / 256, 256, 0, stream>>>(sent, emb, X16);
  k_cvt_pair<<<(2 * G4H * DEMB + 255) / 256, 256, 0, stream>>>(Wihf, Wihb, Wih16, G4H * DEMB);
  k_cvt_pair<<<(2 * G4H * HHID + 255) / 256, 256, 0, stream>>>(Whhf, Whhb, Whh16, G4H * HHID);
  k_cvt_wout<<<(32 * DHID + 255) / 256, 256, 0, stream>>>(Wout, Wout16);

  // input projection: 2*256*64 tiles, 1 wave each, 8 waves/block
  k_gemm_inproj<<<4096, 256, 0, stream>>>(X16, Wih16, bihf, bhhf, bihb, bhhb, XP);

  // sequential scans (both directions concurrently)
  k_lstm_scan<<<2, 256, 0, stream>>>(Whh16, XP, h0, c0, HS);

  // output projection: 256*2 tiles
  k_gemm_feats<<<64, 256, 0, stream>>>(HS, Wout16, bout, feats);

  // Viterbi + backtrace
  k_viterbi<<<1, 32, 0, stream>>>(feats, trans, bptrs, (float*)d_out);
}
